// NativeCatPositionEmbedding_41472204210998
// MI455X (gfx1250) — matrix operand order repaired and checked by reading, not compiled, not run on hardware
//
#include <hip/hip_runtime.h>

// ---------------------------------------------------------------------------
// Fused gather+concat+GEMM for:
//   out[b,p,:] = concat(enc[dfn]+enc[dfn_fa], latent[b,p]) @ W + b
// M = 65536, K = 1024, N = 512.  bf16x3 split-precision WMMA (f32 accum),
// software-pipelined global->LDS staging, NT output stores.
// 512-thread blocks (16 waves, 32x32 per wave) to stay under the ~256-VGPR
// allocator ceiling -> no scratch spills of the prefetch registers.
// ---------------------------------------------------------------------------

typedef __attribute__((ext_vector_type(16))) __bf16 v16bf;
typedef __attribute__((ext_vector_type(8)))  float  v8f;

#define D_MODEL 512
#define K_TOT   1024
#define M_BLK   128
#define N_BLK   128
#define K_BLK   32
#define LDSP    40   // padded LDS row stride in ushorts (80B: conflict-free, 16B aligned)

__device__ __forceinline__ unsigned short f32_to_bf16_rne(float f) {
    unsigned int u = __float_as_uint(f);
    u += 0x7FFFu + ((u >> 16) & 1u);      // round-to-nearest-even
    return (unsigned short)(u >> 16);
}
__device__ __forceinline__ float bf16_to_f32(unsigned short h) {
    return __uint_as_float(((unsigned int)h) << 16);
}

// split 4 floats into hi/lo bf16 pairs, packed 2-per-dword
__device__ __forceinline__ void split_pack4(float x0, float x1, float x2, float x3,
                                            uint2* hi, uint2* lo) {
    unsigned short h0 = f32_to_bf16_rne(x0);
    unsigned short h1 = f32_to_bf16_rne(x1);
    unsigned short h2 = f32_to_bf16_rne(x2);
    unsigned short h3 = f32_to_bf16_rne(x3);
    hi->x = (unsigned int)h0 | ((unsigned int)h1 << 16);
    hi->y = (unsigned int)h2 | ((unsigned int)h3 << 16);
    unsigned short l0 = f32_to_bf16_rne(x0 - bf16_to_f32(h0));
    unsigned short l1 = f32_to_bf16_rne(x1 - bf16_to_f32(h1));
    unsigned short l2 = f32_to_bf16_rne(x2 - bf16_to_f32(h2));
    unsigned short l3 = f32_to_bf16_rne(x3 - bf16_to_f32(h3));
    lo->x = (unsigned int)l0 | ((unsigned int)l1 << 16);
    lo->y = (unsigned int)l2 | ((unsigned int)l3 << 16);
}

// ---------------------------------------------------------------------------
// Kernel 1: W (K x N, f32, row-major) -> Wt_hi / Wt_lo (N x K, bf16)
// ---------------------------------------------------------------------------
__global__ void wprep_kernel(const float* __restrict__ W,
                             unsigned short* __restrict__ wt_hi,
                             unsigned short* __restrict__ wt_lo) {
    int t = blockIdx.x * blockDim.x + threadIdx.x;   // 0 .. N*K-1
    int n = t >> 10;                                 // /K_TOT
    int k = t & (K_TOT - 1);
    float w = W[(size_t)k * D_MODEL + n];
    unsigned short hi = f32_to_bf16_rne(w);
    unsigned short lo = f32_to_bf16_rne(w - bf16_to_f32(hi));
    wt_hi[t] = hi;
    wt_lo[t] = lo;
}

// ---------------------------------------------------------------------------
// Kernel 2: tiled WMMA GEMM, register-prefetch software pipeline
// 512 threads = 16 waves -> 4(M) x 4(N); wave tile 32x32 = 2x2 WMMA tiles
// ---------------------------------------------------------------------------
__global__ __launch_bounds__(512, 1)
void posemb_gemm_kernel(const int*   __restrict__ dfn,
                        const int*   __restrict__ dfn_fa,
                        const float* __restrict__ latent,
                        const float* __restrict__ enc,
                        const unsigned short* __restrict__ wt_hi,
                        const unsigned short* __restrict__ wt_lo,
                        const float* __restrict__ bias,
                        float*       __restrict__ out) {
    __shared__ unsigned short sA_hi[M_BLK * LDSP];
    __shared__ unsigned short sA_lo[M_BLK * LDSP];
    __shared__ unsigned short sB_hi[N_BLK * LDSP];
    __shared__ unsigned short sB_lo[N_BLK * LDSP];

    const int tid = threadIdx.x;
    const int m0  = blockIdx.x * M_BLK;
    const int n0  = blockIdx.y * N_BLK;

    // staging assignment: 4 threads per row, each covering 8 of the 32 K-slice
    const int lr = tid >> 2;        // 0..127
    const int lq = tid & 3;         // 0..3 -> K offset lq*8

    const int rowg = m0 + lr;
    const int i1 = dfn[rowg];
    const int i2 = dfn_fa[rowg];

    // wave tiling: 16 waves -> 4 (M) x 4 (N)
    const int wave = tid >> 5;
    const int lane = tid & 31;
    const int wm   = wave >> 2;     // 0..3
    const int wn   = wave & 3;      // 0..3
    const int l15  = lane & 15;
    const int lhi  = lane >> 4;     // 0/1 (lane half)

    v8f acc[2][2] = {};

    // ---- register prefetch state (raw, no dependent VALU until store phase) ----
    float4 ra[4];          // enc rows (2+2) or latent (2) + zeros
    uint4  rbh, rbl;       // B hi/lo 16B per thread

    auto fetch_tile = [&](int kk) {
        if (kk < D_MODEL) {
            const float4* e1 = (const float4*)(enc + (size_t)i1 * D_MODEL + kk + lq * 8);
            const float4* e2 = (const float4*)(enc + (size_t)i2 * D_MODEL + kk + lq * 8);
#pragma unroll
            for (int i = 0; i < 2; ++i) { ra[i] = e1[i]; ra[2 + i] = e2[i]; }
        } else {
            const float4* L = (const float4*)(latent + (size_t)rowg * D_MODEL
                                              + (kk - D_MODEL) + lq * 8);
#pragma unroll
            for (int i = 0; i < 2; ++i) {
                ra[i] = L[i];
                ra[2 + i] = make_float4(0.f, 0.f, 0.f, 0.f);
            }
        }
        rbh = *(const uint4*)(wt_hi + (size_t)(n0 + lr) * K_TOT + kk + lq * 8);
        rbl = *(const uint4*)(wt_lo + (size_t)(n0 + lr) * K_TOT + kk + lq * 8);
    };

    fetch_tile(0);   // prologue

    for (int kk = 0; kk < K_TOT; kk += K_BLK) {
        // ------------- store phase: split + pack prefetched tile into LDS -------------
#pragma unroll
        for (int i = 0; i < 2; ++i) {
            uint2 hv, lv;
            split_pack4(ra[i].x + ra[2+i].x, ra[i].y + ra[2+i].y,
                        ra[i].z + ra[2+i].z, ra[i].w + ra[2+i].w, &hv, &lv);
            *(uint2*)&sA_hi[lr * LDSP + lq * 8 + i * 4] = hv;
            *(uint2*)&sA_lo[lr * LDSP + lq * 8 + i * 4] = lv;
        }
        *(uint4*)&sB_hi[lr * LDSP + lq * 8] = rbh;
        *(uint4*)&sB_lo[lr * LDSP + lq * 8] = rbl;
        __syncthreads();

        // ------------- issue next tile's global loads (latency hidden by WMMAs) ----
        if (kk + K_BLK < K_TOT) fetch_tile(kk + K_BLK);

        // ------------- compute: 12 WMMAs per wave per K-slice -------------
        union V16 { uint4 u[2]; v16bf v; };
        V16 aH[2], aL[2];
#pragma unroll
        for (int mi = 0; mi < 2; ++mi) {
            // A operand (16x32, 16-bit): lane = row M; K halves split by lane>=16
            int rA = (wm * 32 + mi * 16 + l15) * LDSP + (lhi << 3);
            aH[mi].u[0] = *(const uint4*)&sA_hi[rA];
            aH[mi].u[1] = *(const uint4*)&sA_hi[rA + 16];
            aL[mi].u[0] = *(const uint4*)&sA_lo[rA];
            aL[mi].u[1] = *(const uint4*)&sA_lo[rA + 16];
        }
#pragma unroll
        for (int ni = 0; ni < 2; ++ni) {
            // B operand (32x16, 16-bit): lane = col N; K halves split by lane>=16 (same as A)
            V16 bH, bL;
            int rB = (wn * 32 + ni * 16 + l15) * LDSP + (lhi << 3);
            bH.u[0] = *(const uint4*)&sB_hi[rB];
            bH.u[1] = *(const uint4*)&sB_hi[rB + 16];
            bL.u[0] = *(const uint4*)&sB_lo[rB];
            bL.u[1] = *(const uint4*)&sB_lo[rB + 16];
#pragma unroll
            for (int mi = 0; mi < 2; ++mi) {
                acc[mi][ni] = __builtin_amdgcn_wmma_f32_16x16x32_bf16(
                    false, aH[mi].v, false, bH.v, (short)0, acc[mi][ni], false, false);
                acc[mi][ni] = __builtin_amdgcn_wmma_f32_16x16x32_bf16(
                    false, aH[mi].v, false, bL.v, (short)0, acc[mi][ni], false, false);
                acc[mi][ni] = __builtin_amdgcn_wmma_f32_16x16x32_bf16(
                    false, aL[mi].v, false, bH.v, (short)0, acc[mi][ni], false, false);
            }
        }
        __syncthreads();
    }

    // ------------- epilogue: C/D layout -> global (non-temporal), + bias -------------
#pragma unroll
    for (int ni = 0; ni < 2; ++ni) {
        const int Cg = n0 + wn * 32 + ni * 16 + l15;
        const float bv = bias[Cg];
#pragma unroll
        for (int mi = 0; mi < 2; ++mi) {
            const int Rg = m0 + wm * 32 + mi * 16 + (lhi << 3); // lane half -> M base 0/8
#pragma unroll
            for (int j = 0; j < 8; ++j) {
                __builtin_nontemporal_store(acc[mi][ni][j] + bv,
                                            &out[(size_t)(Rg + j) * D_MODEL + Cg]);
            }
        }
    }
}

// ---------------------------------------------------------------------------
extern "C" void kernel_launch(void* const* d_in, const int* in_sizes, int n_in,
                              void* d_out, int out_size, void* d_ws, size_t ws_size,
                              hipStream_t stream) {
    const int*   dfn    = (const int*)  d_in[0];
    const int*   dfn_fa = (const int*)  d_in[1];
    const float* latent = (const float*)d_in[2];
    const float* enc    = (const float*)d_in[3];
    const float* W      = (const float*)d_in[4];
    const float* bias   = (const float*)d_in[5];
    float*       out    = (float*)d_out;

    unsigned short* wt_hi = (unsigned short*)d_ws;
    unsigned short* wt_lo = wt_hi + (size_t)D_MODEL * K_TOT;

    // 1) transpose + bf16-split W into workspace (2 MB, L2-resident)
    {
        int total = D_MODEL * K_TOT;
        wprep_kernel<<<total / 256, 256, 0, stream>>>(W, wt_hi, wt_lo);
    }

    // 2) fused gather/concat WMMA GEMM
    {
        int m_rows  = in_sizes[0];                 // B*P = 65536
        int mblocks = (m_rows + M_BLK - 1) / M_BLK;
        dim3 grid(mblocks, D_MODEL / N_BLK, 1);    // (512, 4)
        posemb_gemm_kernel<<<grid, 512, 0, stream>>>(dfn, dfn_fa, latent, enc,
                                                     wt_hi, wt_lo, bias, out);
    }
}
